// NAMTuringNoJump_5076651344115
// MI455X (gfx1250) — matrix operand order repaired and checked by reading, not compiled, not run on hardware
//
#include <hip/hip_runtime.h>

// ---------------------------------------------------------------------------
// NAM-Turing (no jump) forward for MI455X / gfx1250 (CDNA5, wave32, WMMA).
//
// - All fp32 GEMMs run on v_wmma_f32_16x16x32_bf16 with a bf16 hi/lo split
//   (3 WMMAs per logical fp32 tile -> near-fp32 accuracy at bf16 rates).
// - GEMM A-panels are staged in LDS via the CDNA5 async global->LDS path
//   (GLOBAL_LOAD_ASYNC_TO_LDS_B128 + s_wait_asynccnt, ASYNCcnt-tracked),
//   double-buffered, one barrier per k-step; falls back to load+ds_store if
//   the async builtins are unavailable in this toolchain.
// - Sequential LSTM: persistent 16-WG kernel, device-scope generation barrier
//   between timesteps; each wave computes all 4 gates of its (batch,col) tile.
// - NTM tape scan: one 64-thread block per (batch,tape); tape column lives
//   entirely in VGPRs (fully unrolled), head positions in LDS.
// ---------------------------------------------------------------------------

typedef __attribute__((ext_vector_type(16))) __bf16 bf16x16;
typedef __attribute__((ext_vector_type(8)))  __bf16 bf16x8;
typedef __attribute__((ext_vector_type(8)))  float  f32x8;
typedef int v4i_ __attribute__((__vector_size__(4 * sizeof(int))));
typedef unsigned short u16;

#define DIM    512
#define NT     8
#define HD     64
#define TLEN   64
#define SEQ    512
#define BATCH  64
#define MROWS  (SEQ * BATCH)      // 32768
#define NACT   (9 * NT)           // 72
#define LSTM_WGS 16

#if __has_builtin(__builtin_amdgcn_global_load_async_to_lds_b128) && \
    __has_builtin(__builtin_amdgcn_s_wait_asynccnt)
#define USE_ASYNC_LDS 1
#else
#define USE_ASYNC_LDS 0
#endif

// ---------------------------------------------------------------- helpers --
__device__ __forceinline__ u16 bfbits(__bf16 x) {
  union { __bf16 b; u16 u; } cv; cv.b = x; return cv.u;
}
__device__ __forceinline__ void split2(float x, u16& h, u16& l) {
  __bf16 bh = (__bf16)x;            // RNE fp32->bf16
  float  r  = x - (float)bh;        // exact residual
  __bf16 bl = (__bf16)r;
  h = bfbits(bh); l = bfbits(bl);
}
__device__ __forceinline__ float sigf(float x) { return 1.0f / (1.0f + __expf(-x)); }

// A fragment straight from global: 16x32 bf16 tile at (m0,k0), row-major lda.
// Lanes 0-15: row m0+L, K = {0..7,16..23}; lanes 16-31: row m0+L-16,
// K = {8..15,24..31}.  (ISA 7.12.2 16-bit A-matrix layout)
__device__ __forceinline__ bf16x16
load_a_frag(const u16* __restrict__ base, int lda, int m0, int k0, int lane) {
  int r  = lane & 15;
  int kb = (lane >> 4) * 8;
  const u16* p = base + (size_t)(m0 + r) * lda + (k0 + kb);
  bf16x8 lo = *(const bf16x8*)(p);
  bf16x8 hi = *(const bf16x8*)(p + 16);
  return __builtin_shufflevector(lo, hi, 0,1,2,3,4,5,6,7,8,9,10,11,12,13,14,15);
}
// B fragment: 32x16 (KxN) where B[k][n] = W[n0+n][k0+k], W row-major [N,ldw].
// Lanes 0-15: col n, K = {0..7,16..23}; lanes 16-31: col n-16, K = {8..15,24..31}.
__device__ __forceinline__ bf16x16
load_b_frag(const u16* __restrict__ W, int ldw, int n0, int k0, int lane, int Nlim) {
  int nl = lane & 15;
  int kb = (lane >> 4) * 8;
  int n  = n0 + nl;
  if (n < Nlim) {
    const u16* p = W + (size_t)n * ldw + (k0 + kb);
    bf16x8 lo = *(const bf16x8*)(p);
    bf16x8 hi = *(const bf16x8*)(p + 16);
    return __builtin_shufflevector(lo, hi, 0,1,2,3,4,5,6,7,8,9,10,11,12,13,14,15);
  }
  bf16x16 z;
#pragma unroll
  for (int i = 0; i < 16; ++i) z[i] = (__bf16)0.0f;
  return z;
}
// Split-bf16 fp32 emulation: c += Ahi*Bhi + Ahi*Blo + Alo*Bhi
__device__ __forceinline__ f32x8
wmma_split(f32x8 c, bf16x16 ah, bf16x16 al, bf16x16 bh, bf16x16 bl) {
  c = __builtin_amdgcn_wmma_f32_16x16x32_bf16(false, ah, false, bh, (short)0, c, false, false);
  c = __builtin_amdgcn_wmma_f32_16x16x32_bf16(false, ah, false, bl, (short)0, c, false, false);
  c = __builtin_amdgcn_wmma_f32_16x16x32_bf16(false, al, false, bh, (short)0, c, false, false);
  return c;
}

// 16-byte global -> LDS staging (async DMA on CDNA5, sync fallback otherwise).
// Builtin parameter types (from hipcc diagnostic): int4-vector pointers in
// AS1 (printed as __device__) and AS3.
__device__ __forceinline__ void stage16(const u16* __restrict__ g, u16* l) {
#if USE_ASYNC_LDS
  __builtin_amdgcn_global_load_async_to_lds_b128(
      (__attribute__((address_space(1))) v4i_*)(uintptr_t)g,
      (__attribute__((address_space(3))) v4i_*)l, 0, 0);
#else
  *(uint4*)l = *(const uint4*)g;
#endif
}
__device__ __forceinline__ void stage_wait() {
#if USE_ASYNC_LDS
  __builtin_amdgcn_s_wait_asynccnt(0);
#endif
}

// ------------------------------------------------------------ prep kernels --
__global__ void cvt_pair_kernel(const float* __restrict__ x,
                                u16* __restrict__ hi, u16* __restrict__ lo, int n) {
  int i = blockIdx.x * blockDim.x + threadIdx.x;
  if (i < n) { u16 h, l; split2(x[i], h, l); hi[i] = h; lo[i] = l; }
}
__global__ void init_state_kernel(float* __restrict__ cstate, unsigned* __restrict__ sync) {
  int i = blockIdx.x * blockDim.x + threadIdx.x;
  if (i < BATCH * DIM) cstate[i] = 0.0f;
  if (i == 0) *sync = 0u;
}

// --------------------------------------------------------- generic GEMM -----
// C[M,N] = Apair[M,K] @ Wpair[N,K]^T + bias.  Block = 8 waves, block tile is
// 64(M) x 128(N); each wave owns a 64x16 strip (4 accum tiles).
// The shared 64x32 A panel (hi+lo) is staged in LDS once per k-step via the
// async global->LDS path, double-buffered (one barrier per k-step).
// LDS row stride 56 elems = 112 B: 16B-aligned b128 fragments, 28*r mod 64
// bank pattern is conflict-free over the 16 rows a fragment touches.
#define A_RS 56
__global__ void __launch_bounds__(256)
gemm_bf16pair(const u16* __restrict__ Ah, const u16* __restrict__ Al,
              const u16* __restrict__ Wh, const u16* __restrict__ Wl,
              const float* __restrict__ bias, float* __restrict__ C,
              int M, int N, int K) {
  __shared__ u16 sA[2][2][64 * A_RS];          // [buf][hi/lo][row*A_RS + k]
  const int lane = threadIdx.x & 31;
  const int wave = threadIdx.x >> 5;
  const int n0 = blockIdx.x * 128 + wave * 16;
  const int m0 = blockIdx.y * 64;
  const bool active = (n0 < N);
  const int srow = threadIdx.x >> 2;           // 0..63: one 16B chunk/thread
  const int sch  = (threadIdx.x & 3) * 8;      // elem offset 0,8,16,24

  f32x8 acc[4] = {};
  for (int k0 = 0, it = 0; k0 < K; k0 += 32, ++it) {
    const int buf = it & 1;
    // cooperative A-panel staging (all 256 threads, hi and lo)
    stage16(Ah + (size_t)(m0 + srow) * K + k0 + sch, &sA[buf][0][srow * A_RS + sch]);
    stage16(Al + (size_t)(m0 + srow) * K + k0 + sch, &sA[buf][1][srow * A_RS + sch]);
    bf16x16 bh, bl;
    if (active) {                               // per-wave B fragments
      bh = load_b_frag(Wh, K, n0, k0, lane, N);
      bl = load_b_frag(Wl, K, n0, k0, lane, N);
    }
    stage_wait();                               // own wave's async stores done
    __syncthreads();                            // panel visible to all waves
    if (active) {
      const int r  = lane & 15;
      const int kb = (lane >> 4) * 8;
#pragma unroll
      for (int mi = 0; mi < 4; ++mi) {
        const u16* ph = &sA[buf][0][(mi * 16 + r) * A_RS + kb];
        const u16* pl = &sA[buf][1][(mi * 16 + r) * A_RS + kb];
        bf16x8 h0 = *(const bf16x8*)(ph);
        bf16x8 h1 = *(const bf16x8*)(ph + 16);
        bf16x8 l0 = *(const bf16x8*)(pl);
        bf16x8 l1 = *(const bf16x8*)(pl + 16);
        bf16x16 ahf = __builtin_shufflevector(h0, h1, 0,1,2,3,4,5,6,7,8,9,10,11,12,13,14,15);
        bf16x16 alf = __builtin_shufflevector(l0, l1, 0,1,2,3,4,5,6,7,8,9,10,11,12,13,14,15);
        acc[mi] = wmma_split(acc[mi], ahf, alf, bh, bl);
      }
    }
    // no trailing barrier: buffer `buf` is overwritten at it+2, and the
    // barrier at it+1 orders that against this iteration's ds reads.
  }
  if (!active) return;
  const int col = n0 + (lane & 15);
  if (col >= N) return;
  const float bv = bias ? bias[col] : 0.0f;
  const int rb = (lane >> 4) << 3;              // C/D layout: VGPR v -> M=v(+8)
#pragma unroll
  for (int mi = 0; mi < 4; ++mi)
#pragma unroll
    for (int v = 0; v < 8; ++v)
      C[(size_t)(m0 + mi * 16 + rb + v) * N + col] = acc[mi][v] + bv;
}

// ---------------------------------------------------------------- LSTM ------
// Persistent kernel: 16 WGs x 256 threads.  WG wg owns hidden columns
// [32*wg, 32*wg+32).  Wave (mi=wave>>1, ct=wave&1) computes ALL FOUR gates
// for batch-tile mi and col-tile ct, so i/f/g/o for (b,j) coexist per-thread.
// Device-scope generation barrier separates timesteps.
__global__ void __launch_bounds__(256)
lstm_kernel(const u16* __restrict__ Xh, const u16* __restrict__ Xl,
            const u16* __restrict__ Wih_h, const u16* __restrict__ Wih_l,
            const u16* __restrict__ Whh_h, const u16* __restrict__ Whh_l,
            const float* __restrict__ b_ih, const float* __restrict__ b_hh,
            u16* __restrict__ Hh, u16* __restrict__ Hl,
            float* __restrict__ cstate, unsigned* __restrict__ sync) {
  const int lane = threadIdx.x & 31;
  const int wave = threadIdx.x >> 5;
  const int mi = wave >> 1;                     // batch tile 0..3
  const int ct = wave & 1;                      // col tile 0..1
  const int jt = blockIdx.x * 32 + ct * 16;     // hidden column tile base
  const int m0 = mi * 16;
  const int j  = jt + (lane & 15);

  const float bi = b_ih[j]           + b_hh[j];
  const float bf = b_ih[DIM + j]     + b_hh[DIM + j];
  const float bg = b_ih[2 * DIM + j] + b_hh[2 * DIM + j];
  const float bo = b_ih[3 * DIM + j] + b_hh[3 * DIM + j];

  for (int t = 0; t < SEQ; ++t) {
    f32x8 acc[4] = {};                          // gates i,f,g,o
    const u16* Axh = Xh + (size_t)t * BATCH * DIM;
    const u16* Axl = Xl + (size_t)t * BATCH * DIM;
    for (int k0 = 0; k0 < DIM; k0 += 32) {      // x_t @ W_ih^T
      bf16x16 ah = load_a_frag(Axh, DIM, m0, k0, lane);
      bf16x16 al = load_a_frag(Axl, DIM, m0, k0, lane);
#pragma unroll
      for (int g = 0; g < 4; ++g) {
        bf16x16 bh = load_b_frag(Wih_h, DIM, g * DIM + jt, k0, lane, 4 * DIM);
        bf16x16 bl = load_b_frag(Wih_l, DIM, g * DIM + jt, k0, lane, 4 * DIM);
        acc[g] = wmma_split(acc[g], ah, al, bh, bl);
      }
    }
    if (t > 0) {                                // h_{t-1} @ W_hh^T
      const u16* Ahh = Hh + (size_t)(t - 1) * BATCH * DIM;
      const u16* Ahl = Hl + (size_t)(t - 1) * BATCH * DIM;
      for (int k0 = 0; k0 < DIM; k0 += 32) {
        bf16x16 ah = load_a_frag(Ahh, DIM, m0, k0, lane);
        bf16x16 al = load_a_frag(Ahl, DIM, m0, k0, lane);
#pragma unroll
        for (int g = 0; g < 4; ++g) {
          bf16x16 bh = load_b_frag(Whh_h, DIM, g * DIM + jt, k0, lane, 4 * DIM);
          bf16x16 bl = load_b_frag(Whh_l, DIM, g * DIM + jt, k0, lane, 4 * DIM);
          acc[g] = wmma_split(acc[g], ah, al, bh, bl);
        }
      }
    }
    const int rb = (lane >> 4) << 3;
#pragma unroll
    for (int v = 0; v < 8; ++v) {
      const int b = m0 + rb + v;
      float gi = acc[0][v] + bi;
      float gf = acc[1][v] + bf;
      float gg = acc[2][v] + bg;
      float go = acc[3][v] + bo;
      float c_old = cstate[b * DIM + j];
      float c_new = sigf(gf) * c_old + sigf(gi) * tanhf(gg);
      float h     = sigf(go) * tanhf(c_new);
      cstate[b * DIM + j] = c_new;
      size_t hidx = (size_t)(t * BATCH + b) * DIM + j;
      u16 hh, hl; split2(h, hh, hl);
      Hh[hidx] = hh; Hl[hidx] = hl;
    }
    // ---- device-scope generation barrier between timesteps ----
    __threadfence();
    __syncthreads();
    if (threadIdx.x == 0) {
      atomicAdd(sync, 1u);
      const unsigned target = (unsigned)(t + 1) * (unsigned)gridDim.x;
      while (__hip_atomic_load(sync, __ATOMIC_ACQUIRE, __HIP_MEMORY_SCOPE_AGENT) < target)
        __builtin_amdgcn_s_sleep(2);
    }
    __syncthreads();
  }
}

// ---------------------------------------------------------------- tape ------
// One 64-thread block per (tape tp, batch n).  Thread c owns tape column c:
// 64 fp32 values fully register-resident (unrolled l-loops).  Head positions
// live in LDS (broadcast reads).  read_outs emitted as bf16 pair for the
// final WMMA GEMM; final tape written straight to d_out's second region.
__global__ void __launch_bounds__(64)
tape_kernel(const float* __restrict__ values, const float* __restrict__ actions,
            u16* __restrict__ ROh, u16* __restrict__ ROl,
            float* __restrict__ tape_out) {
  const int tp = blockIdx.x;       // 0..7
  const int n  = blockIdx.y;       // 0..63
  const int c  = threadIdx.x;      // 0..63
  __shared__ float wpos[64], rpos[64], scal[9];
  float tape[TLEN];
#pragma unroll
  for (int l = 0; l < TLEN; ++l) tape[l] = 0.0f;
  wpos[c] = (c == 0) ? 1.0f : 0.0f;
  rpos[c] = (c == 0) ? 1.0f : 0.0f;
  __syncthreads();

  for (int s = 0; s < SEQ; ++s) {
    if (c == 0) {
      const float* a = actions + (size_t)(s * BATCH + n) * NACT + tp * 9;
      float m1 = fmaxf(a[0], fmaxf(a[1], a[2]));
      float e0 = __expf(a[0] - m1), e1 = __expf(a[1] - m1), e2 = __expf(a[2] - m1);
      float is = 1.0f / (e0 + e1 + e2);
      scal[0] = e0 * is; scal[1] = e1 * is; scal[2] = e2 * is;
      float m2 = fmaxf(a[3], fmaxf(a[4], a[5]));
      float f0 = __expf(a[3] - m2), f1 = __expf(a[4] - m2), f2 = __expf(a[5] - m2);
      float js = 1.0f / (f0 + f1 + f2);
      scal[3] = f0 * js; scal[4] = f1 * js; scal[5] = f2 * js;
      scal[6] = sigf(a[6]); scal[7] = sigf(a[7]); scal[8] = sigf(a[8]);
    }
    __syncthreads();
    const float rd0 = scal[0], rd1 = scal[1], rd2 = scal[2];
    const float wd0 = scal[3], wd1 = scal[4], wd2 = scal[5];
    const float rw0 = scal[6], rw1 = scal[7], rw2 = scal[8];
    const float v = values[(size_t)(s * BATCH + n) * DIM + tp * HD + c];

    float oldval = 0.0f;                        // sum_l tape[l][c] * wpos[l]
#pragma unroll
    for (int l = 0; l < TLEN; ++l) oldval += tape[l] * wpos[l];
    const float upd = v * rw1 - oldval * rw2;   // write - erase
#pragma unroll
    for (int l = 0; l < TLEN; ++l) tape[l] += wpos[l] * upd;

    // rolls read OLD positions; read_out reads NEW tape, OLD rpos.
    const float nw = wpos[(c + 1) & 63] * wd0 + wpos[c] * wd1 + wpos[(c + 63) & 63] * wd2;
    float ro = 0.0f;
#pragma unroll
    for (int l = 0; l < TLEN; ++l) ro += tape[l] * rpos[l];
    ro *= rw0;
    const float nr = rpos[(c + 1) & 63] * rd0 + rpos[c] * rd1 + rpos[(c + 63) & 63] * rd2;
    __syncthreads();
    wpos[c] = nw; rpos[c] = nr;

    size_t oi = (size_t)(s * BATCH + n) * DIM + tp * HD + c;
    u16 rh, rl; split2(ro, rh, rl);
    ROh[oi] = rh; ROl[oi] = rl;
    __syncthreads();
  }
#pragma unroll
  for (int l = 0; l < TLEN; ++l)
    tape_out[(size_t)(l * BATCH + n) * DIM + tp * HD + c] = tape[l];
}

// ---------------------------------------------------------------- launch ----
extern "C" void kernel_launch(void* const* d_in, const int* in_sizes, int n_in,
                              void* d_out, int out_size, void* d_ws, size_t ws_size,
                              hipStream_t stream) {
  (void)in_sizes; (void)n_in; (void)out_size; (void)ws_size;
  const float* inputs = (const float*)d_in[0];
  const float* W_ih   = (const float*)d_in[1];
  const float* W_hh   = (const float*)d_in[2];
  const float* b_ih   = (const float*)d_in[3];
  const float* b_hh   = (const float*)d_in[4];
  const float* W_act  = (const float*)d_in[5];
  const float* b_act  = (const float*)d_in[6];
  const float* W_val  = (const float*)d_in[7];
  const float* b_val  = (const float*)d_in[8];
  const float* W_out  = (const float*)d_in[9];
  const float* b_out  = (const float*)d_in[10];

  char* ws = (char*)d_ws;
  size_t off = 0;
  auto take = [&](size_t bytes) -> char* {
    char* p = ws + off;
    off += (bytes + 255) & ~(size_t)255;
    return p;
  };
  const size_t nX   = (size_t)MROWS * DIM;        // 16.8M
  const size_t nWih = (size_t)4 * DIM * DIM;      // 1.05M
  const size_t nWv  = (size_t)NT * HD * DIM;      // 262K
  const size_t nWa  = (size_t)NACT * DIM;         // 36.9K
  u16* Xh    = (u16*)take(nX * 2);    u16* Xl    = (u16*)take(nX * 2);
  u16* Wih_h = (u16*)take(nWih * 2);  u16* Wih_l = (u16*)take(nWih * 2);
  u16* Whh_h = (u16*)take(nWih * 2);  u16* Whh_l = (u16*)take(nWih * 2);
  u16* Wv_h  = (u16*)take(nWv * 2);   u16* Wv_l  = (u16*)take(nWv * 2);
  u16* Wa_h  = (u16*)take(nWa * 2);   u16* Wa_l  = (u16*)take(nWa * 2);
  u16* Wo_h  = (u16*)take(nWv * 2);   u16* Wo_l  = (u16*)take(nWv * 2);
  u16* Hh    = (u16*)take(nX * 2);    u16* Hl    = (u16*)take(nX * 2);
  u16* ROh   = (u16*)take(nX * 2);    u16* ROl   = (u16*)take(nX * 2);
  float* values  = (float*)take(nX * 4);
  float* actions = (float*)take((size_t)MROWS * NACT * 4);
  float* cstate  = (float*)take((size_t)BATCH * DIM * 4);
  unsigned* sync = (unsigned*)take(256);

  float* out_main = (float*)d_out;                 // [SEQ,BATCH,DIM]
  float* out_tape = out_main + nX;                 // [TLEN,BATCH,DIM]

  // 0) fp32 -> bf16 hi/lo pairs
  auto cvt = [&](const float* src, u16* hi, u16* lo, size_t n) {
    cvt_pair_kernel<<<(unsigned)((n + 255) / 256), 256, 0, stream>>>(src, hi, lo, (int)n);
  };
  cvt(inputs, Xh, Xl, nX);
  cvt(W_ih, Wih_h, Wih_l, nWih);
  cvt(W_hh, Whh_h, Whh_l, nWih);
  cvt(W_val, Wv_h, Wv_l, nWv);
  cvt(W_act, Wa_h, Wa_l, nWa);
  cvt(W_out, Wo_h, Wo_l, nWv);
  init_state_kernel<<<(BATCH * DIM + 255) / 256, 256, 0, stream>>>(cstate, sync);

  // 1) values = inputs @ W_val^T + b_val          [32768, 512]
  gemm_bf16pair<<<dim3(DIM / 128, MROWS / 64), 256, 0, stream>>>(
      Xh, Xl, Wv_h, Wv_l, b_val, values, MROWS, DIM, DIM);

  // 2) sequential LSTM -> hidden (bf16 pair)
  lstm_kernel<<<LSTM_WGS, 256, 0, stream>>>(
      Xh, Xl, Wih_h, Wih_l, Whh_h, Whh_l, b_ih, b_hh, Hh, Hl, cstate, sync);

  // 3) actions = hidden @ W_act^T + b_act         [32768, 72]
  gemm_bf16pair<<<dim3(1, MROWS / 64), 256, 0, stream>>>(
      Hh, Hl, Wa_h, Wa_l, b_act, actions, MROWS, NACT, DIM);

  // 4) tape scan -> read_outs (bf16 pair) + final tape into d_out
  tape_kernel<<<dim3(NT, BATCH), 64, 0, stream>>>(values, actions, ROh, ROl, out_tape);

  // 5) outputs = read_outs @ W_out^T + b_out -> d_out [32768, 512]
  gemm_bf16pair<<<dim3(DIM / 128, MROWS / 64), 256, 0, stream>>>(
      ROh, ROl, Wo_h, Wo_l, b_out, out_main, MROWS, DIM, DIM);
}
